// LinearSelfAttention_12249246728267
// MI455X (gfx1250) — matrix operand order repaired and checked by reading, not compiled, not run on hardware
//
#include <hip/hip_runtime.h>

typedef __attribute__((ext_vector_type(16))) __bf16 v16bf;
typedef __attribute__((ext_vector_type(8)))  float  v8f;

union Frag {
  uint4 q2[2];
  unsigned short u[16];
  v16bf v;
};

__device__ __forceinline__ unsigned int pack2bf(float a, float b) {
  unsigned int ua = __float_as_uint(a) + 0x8000u;
  unsigned int ub = __float_as_uint(b) + 0x8000u;
  return __builtin_amdgcn_perm(ub, ua, 0x07060302u);  // {hi16(b), hi16(a)}
}
__device__ __forceinline__ unsigned short f2bf(float f) {
  return (unsigned short)((__float_as_uint(f) + 0x8000u) >> 16);
}
__device__ __forceinline__ float bf2f(unsigned short h) {
  return __uint_as_float(((unsigned int)h) << 16);
}
#define WMMA_BF16(A, B, C) \
  __builtin_amdgcn_wmma_f32_16x16x32_bf16(false, (A), false, (B), (short)0, (C), false, false)

// CDNA5 async global->LDS copy (ASYNCcnt-tracked), ISA §15.18 opcode 98.
// LDS operand = LDS byte offset (low 32 bits of generic LDS address).
__device__ __forceinline__ void async_b128(unsigned short* lds, const unsigned short* g) {
  unsigned int lo = (unsigned int)(size_t)lds;
  asm volatile("global_load_async_to_lds_b128 %0, %1, off" :: "v"(lo), "v"(g) : "memory");
}
__device__ __forceinline__ void wait_async() {
#if __has_builtin(__builtin_amdgcn_s_wait_asynccnt)
  __builtin_amdgcn_s_wait_asynccnt(0);
#else
  asm volatile("s_wait_asynccnt 0x0" ::: "memory");
#endif
}

// padded LDS strides (u16 units; rows stay 16B-aligned, conflict-free-ish)
#define LDA_S 40    // A tile rows: 32 data + 8 pad  (80B -> bank 20*l, distinct)
#define LDB_S 136   // B tile rows: 128 data + 8 pad (272B -> 2-way worst case)
#define OST_S 136   // output staging rows: 128 data + 8 pad
#define LS2   72    // kernel-2 64x64 tiles: 64 data + 8 pad (144B)

#define A_BUF (128 * LDA_S)
#define B_BUF (32 * LDB_S)
#define B_OFF (2 * A_BUF)

// -----------------------------------------------------------------------------
// Kernel 0: one-shot f32 -> bf16 convert (x and the four weight matrices).
// Each thread: 8 floats -> 8 bf16 (two float4 loads, one uint4 store).
// -----------------------------------------------------------------------------
__global__ __launch_bounds__(256) void cvt_bf16_kernel(
    const float* __restrict__ src, unsigned short* __restrict__ dst, int n8) {
  int i = blockIdx.x * 256 + threadIdx.x;
  if (i >= n8) return;
  float4 a = *(const float4*)(src + (size_t)i * 8);
  float4 b = *(const float4*)(src + (size_t)i * 8 + 4);
  uint4 o;
  o.x = pack2bf(a.x, a.y); o.y = pack2bf(a.z, a.w);
  o.z = pack2bf(b.x, b.y); o.w = pack2bf(b.z, b.w);
  *(uint4*)(dst + (size_t)i * 8) = o;
}

// B=4, T=4096, DIM=1024, H=16, E=64, BUCKET=64, U=64
// -----------------------------------------------------------------------------
// Kernel 1: q/k/v = x @ W{q,k,v} (all bf16 in), fused softmax / elu+1,
// bf16 out [b,h,t,e]. 256 threads, 128x128 tile, K-chunk 32, double-buffered
// LDS filled by async global->LDS copies. grid = (128, 8, 3)
// -----------------------------------------------------------------------------
__global__ __launch_bounds__(256) void qkv_gemm_kernel(
    const unsigned short* __restrict__ xb, const unsigned short* __restrict__ wqb,
    const unsigned short* __restrict__ wkb, const unsigned short* __restrict__ wvb,
    unsigned short* __restrict__ qout, unsigned short* __restrict__ kout,
    unsigned short* __restrict__ vout) {
  __shared__ __align__(16) unsigned short sm[2 * A_BUF + 2 * B_BUF];

  const int tid = threadIdx.x;
  const int w = tid >> 5, l = tid & 31;
  const int wm = w & 3, wn = w >> 2;     // wave -> 32-row group x 64-col group
  const int mtile = blockIdx.x * 128;
  const int ntile = blockIdx.y * 128;
  const int z = blockIdx.z;              // 0=q 1=k 2=v
  const unsigned short* Wb = (z == 0) ? wqb : (z == 1) ? wkb : wvb;
  const int hb = (l >> 4) << 3;
  const int lane15 = l & 15;

  const int arow = tid >> 1, acg = (tid & 1) * 16;  // A staging: 128x32
  const int brow = tid >> 3, bcg = (tid & 7) * 16;  // B staging: 32x128

  v8f acc[2][4];
#pragma unroll
  for (int s = 0; s < 2; ++s)
#pragma unroll
    for (int t = 0; t < 4; ++t)
#pragma unroll
      for (int i = 0; i < 8; ++i) acc[s][t][i] = 0.0f;

  const unsigned short* aptr = xb + (size_t)(mtile + arow) * 1024 + acg;
  const unsigned short* bptr = Wb + (size_t)brow * 1024 + ntile + bcg;
  unsigned short* const lAs = sm + arow * LDA_S + acg;
  unsigned short* const lBs = sm + B_OFF + brow * LDB_S + bcg;

  // prologue: async-stage chunk 0 into buffer 0
  async_b128(lAs, aptr);      async_b128(lAs + 8, aptr + 8);
  async_b128(lBs, bptr);      async_b128(lBs + 8, bptr + 8);
  wait_async();
  __syncthreads();

  int cur = 0;
  for (int kc = 0; kc < 1024; kc += 32) {
    const bool more = (kc + 32) < 1024;
    unsigned short* lAc = sm + cur * A_BUF;
    unsigned short* lBc = sm + B_OFF + cur * B_BUF;
    if (more) {  // async-stage next chunk into the other buffer
      const unsigned short* as = aptr + (kc + 32);
      const unsigned short* bs = bptr + (size_t)(kc + 32) * 1024;
      unsigned short* ad = lAs + (cur ^ 1) * A_BUF;
      unsigned short* bd = lBs + (cur ^ 1) * B_BUF;
      async_b128(ad, as);     async_b128(ad + 8, as + 8);
      async_b128(bd, bs);     async_b128(bd + 8, bs + 8);
      if (kc + 64 < 1024) {
        __builtin_prefetch(aptr + (kc + 64), 0, 0);
        __builtin_prefetch(bptr + (size_t)(kc + 64) * 1024, 0, 0);
      }
    }
    Frag a0, a1;
    {
      const unsigned short* ap = lAc + (wm * 32 + lane15) * LDA_S + hb;
      a0.q2[0] = *(const uint4*)(ap);
      a0.q2[1] = *(const uint4*)(ap + 16);
      const unsigned short* ap2 = ap + 16 * LDA_S;
      a1.q2[0] = *(const uint4*)(ap2);
      a1.q2[1] = *(const uint4*)(ap2 + 16);
    }
#pragma unroll
    for (int t = 0; t < 4; ++t) {
      Frag bfr;
      const unsigned short* bp = lBc + wn * 64 + t * 16 + lane15;
#pragma unroll
      for (int i = 0; i < 8; ++i) {
        bfr.u[i]     = bp[(hb + i) * LDB_S];
        bfr.u[8 + i] = bp[(hb + 16 + i) * LDB_S];
      }
      acc[0][t] = WMMA_BF16(a0.v, bfr.v, acc[0][t]);
      acc[1][t] = WMMA_BF16(a1.v, bfr.v, acc[1][t]);
    }
    if (more) wait_async();
    __syncthreads();
    cur ^= 1;
  }

  // ---- fused epilogue (wave's 64 cols == one full head) ----
  if (z == 0) {  // softmax over the head's 64 features per row
#pragma unroll
    for (int s = 0; s < 2; ++s)
#pragma unroll
      for (int r = 0; r < 8; ++r) {
        float m0 = fmaxf(fmaxf(acc[s][0][r], acc[s][1][r]),
                         fmaxf(acc[s][2][r], acc[s][3][r]));
        m0 = fmaxf(m0, __shfl_xor(m0, 1, 32));
        m0 = fmaxf(m0, __shfl_xor(m0, 2, 32));
        m0 = fmaxf(m0, __shfl_xor(m0, 4, 32));
        m0 = fmaxf(m0, __shfl_xor(m0, 8, 32));
        float sum = 0.0f;
#pragma unroll
        for (int t = 0; t < 4; ++t) {
          float e = __expf(acc[s][t][r] - m0);
          acc[s][t][r] = e; sum += e;
        }
        sum += __shfl_xor(sum, 1, 32); sum += __shfl_xor(sum, 2, 32);
        sum += __shfl_xor(sum, 4, 32); sum += __shfl_xor(sum, 8, 32);
        float inv = 1.0f / sum;
#pragma unroll
        for (int t = 0; t < 4; ++t) acc[s][t][r] *= inv;
      }
  } else if (z == 1) {  // elu(k)+1
#pragma unroll
    for (int s = 0; s < 2; ++s)
#pragma unroll
      for (int t = 0; t < 4; ++t)
#pragma unroll
        for (int r = 0; r < 8; ++r) {
          float v = acc[s][t][r];
          acc[s][t][r] = (v > 0.0f) ? (v + 1.0f) : __expf(v);
        }
  }
  // stage full 128x128 bf16 tile, then coalesced 128B-row stores per head
  unsigned short* ostage = sm;  // 128 x OST_S (fits in sm)
#pragma unroll
  for (int s = 0; s < 2; ++s)
#pragma unroll
    for (int r = 0; r < 8; ++r) {
      int row = wm * 32 + s * 16 + r + hb;
#pragma unroll
      for (int t = 0; t < 4; ++t)
        ostage[row * OST_S + wn * 64 + t * 16 + lane15] = f2bf(acc[s][t][r]);
    }
  __syncthreads();
  unsigned short* outp = (z == 0) ? qout : (z == 1) ? kout : vout;
  {
    int row = tid >> 1, seg = tid & 1;  // 128 rows x 2 head-segments
    int m = mtile + row, bb = m >> 12, tt = m & 4095;
    int head = blockIdx.y * 2 + seg;
    uint4* dst = (uint4*)(outp + (((size_t)(bb * 16 + head)) * 4096 + tt) * 64);
    const uint4* src = (const uint4*)(ostage + row * OST_S + seg * 64);
#pragma unroll
    for (int j = 0; j < 8; ++j) dst[j] = src[j];
  }
}

// -----------------------------------------------------------------------------
// Kernel 2: bucketed causal linear attention, one (b,h) per block.
// grid = (64), block = 128 (4 waves). Running ctx in wave accumulators;
// k/v bucket tiles staged with async global->LDS.
// -----------------------------------------------------------------------------
__global__ __launch_bounds__(128) void attn_kernel(
    const unsigned short* __restrict__ qb, const unsigned short* __restrict__ kb,
    const unsigned short* __restrict__ vb, unsigned short* __restrict__ attnb) {
  __shared__ __align__(16) unsigned short kls[64 * LS2];
  __shared__ __align__(16) unsigned short vls[64 * LS2];
  __shared__ __align__(16) unsigned short cls[64 * LS2];
  __shared__ float ksum[64];
  const int tid = threadIdx.x, w = tid >> 5, l = tid & 31;
  const int bh = blockIdx.x, b = bh >> 4, h = bh & 15;
  const size_t headbase = (size_t)bh * 4096 * 64;
  const int hb = (l >> 4) << 3;
  const int lane15 = l & 15;

  v8f ctx[4];
#pragma unroll
  for (int t = 0; t < 4; ++t)
#pragma unroll
    for (int i = 0; i < 8; ++i) ctx[t][i] = 0.0f;
  if (tid < 64) ksum[tid] = 0.0f;
  __syncthreads();

  const int srow = tid >> 1, shalf = (tid & 1) * 32;  // bucket staging coords

  for (int u = 0; u < 64; ++u) {
    const size_t tokbase = headbase + (size_t)u * 64 * 64;
    {  // async-stage k,v bucket tiles (padded LDS rows)
      const unsigned short* ks = kb + tokbase + srow * 64 + shalf;
      const unsigned short* vs = vb + tokbase + srow * 64 + shalf;
      unsigned short* kd = kls + srow * LS2 + shalf;
      unsigned short* vd = vls + srow * LS2 + shalf;
#pragma unroll
      for (int j = 0; j < 4; ++j) {
        async_b128(kd + j * 8, ks + j * 8);
        async_b128(vd + j * 8, vs + j * 8);
      }
    }
    {  // step 1: normalized (pre-update => blindspot shift) ctx -> cls
#pragma unroll
      for (int r = 0; r < 8; ++r) {
        int d = w * 16 + r + hb;
        float inv = 1.0f / (ksum[d] + 1e-6f);
#pragma unroll
        for (int t = 0; t < 4; ++t)
          cls[d * LS2 + t * 16 + lane15] = f2bf(ctx[t][r] * inv);
      }
    }
    wait_async();
    __syncthreads();
    {  // step 2: attn_u = q_u @ cls
      v8f aacc[4];
#pragma unroll
      for (int t = 0; t < 4; ++t)
#pragma unroll
        for (int i = 0; i < 8; ++i) aacc[t][i] = 0.0f;
#pragma unroll
      for (int kc = 0; kc < 64; kc += 32) {
        Frag a;
        {
          int tok = u * 64 + w * 16 + lane15;
          const unsigned short* qr = qb + headbase + (size_t)tok * 64 + kc + hb;
          a.q2[0] = *(const uint4*)(qr);
          a.q2[1] = *(const uint4*)(qr + 16);
        }
#pragma unroll
        for (int t = 0; t < 4; ++t) {
          Frag bfr;
          const unsigned short* bp = cls + kc * LS2 + t * 16 + lane15;
#pragma unroll
          for (int i = 0; i < 8; ++i) {
            bfr.u[i]     = bp[(hb + i) * LS2];
            bfr.u[8 + i] = bp[(hb + 16 + i) * LS2];
          }
          aacc[t] = WMMA_BF16(a.v, bfr.v, aacc[t]);
        }
      }
#pragma unroll
      for (int r = 0; r < 8; ++r) {  // attn bf16 -> [b, t, dim]
        int tok = u * 64 + w * 16 + r + hb;
        size_t obase = ((size_t)b * 4096 + tok) * 1024 + h * 64;
#pragma unroll
        for (int t = 0; t < 4; ++t)
          attnb[obase + t * 16 + lane15] = f2bf(aacc[t][r]);
      }
    }
    __syncthreads();  // cls reads done; kls/vls ready for consumers
    if (tid < 64) {   // step 4: ksum += column sums of k bucket
      float s = 0.0f;
      for (int n = 0; n < 64; ++n) s += bf2f(kls[n * LS2 + tid]);
      ksum[tid] += s;
    }
    {  // step 5: ctx += k_u^T @ v_u
#pragma unroll
      for (int kc = 0; kc < 64; kc += 32) {
        Frag a;  // A[m=d][kk=n] = k[n][d] (transposed gather)
        {
          int d = w * 16 + lane15;
          int nb = kc + hb;
#pragma unroll
          for (int i = 0; i < 8; ++i) {
            a.u[i]     = kls[(nb + i) * LS2 + d];
            a.u[8 + i] = kls[(nb + 16 + i) * LS2 + d];
          }
        }
#pragma unroll
        for (int t = 0; t < 4; ++t) {
          Frag bfr;
          const unsigned short* bp = vls + kc * LS2 + t * 16 + lane15;
#pragma unroll
          for (int i = 0; i < 8; ++i) {
            bfr.u[i]     = bp[(hb + i) * LS2];
            bfr.u[8 + i] = bp[(hb + 16 + i) * LS2];
          }
          ctx[t] = WMMA_BF16(a.v, bfr.v, ctx[t]);
        }
      }
    }
    __syncthreads();
  }
}

// -----------------------------------------------------------------------------
// Kernel 3: out = attn @ Wo + bo (f32 out). Same async-staged tiling.
// grid = (128, 8)
// -----------------------------------------------------------------------------
__global__ __launch_bounds__(256) void out_gemm_kernel(
    const unsigned short* __restrict__ attnb, const unsigned short* __restrict__ wob,
    const float* __restrict__ bo, float* __restrict__ out) {
  __shared__ __align__(16) unsigned short sm[2 * A_BUF + 2 * B_BUF];

  const int tid = threadIdx.x;
  const int w = tid >> 5, l = tid & 31;
  const int wm = w & 3, wn = w >> 2;
  const int mtile = blockIdx.x * 128;
  const int ntile = blockIdx.y * 128;
  const int hb = (l >> 4) << 3;
  const int lane15 = l & 15;

  const int arow = tid >> 1, acg = (tid & 1) * 16;
  const int brow = tid >> 3, bcg = (tid & 7) * 16;

  v8f acc[2][4];
#pragma unroll
  for (int s = 0; s < 2; ++s)
#pragma unroll
    for (int t = 0; t < 4; ++t)
#pragma unroll
      for (int i = 0; i < 8; ++i) acc[s][t][i] = 0.0f;

  const unsigned short* aptr = attnb + (size_t)(mtile + arow) * 1024 + acg;
  const unsigned short* bptr = wob + (size_t)brow * 1024 + ntile + bcg;
  unsigned short* const lAs = sm + arow * LDA_S + acg;
  unsigned short* const lBs = sm + B_OFF + brow * LDB_S + bcg;

  async_b128(lAs, aptr);      async_b128(lAs + 8, aptr + 8);
  async_b128(lBs, bptr);      async_b128(lBs + 8, bptr + 8);
  wait_async();
  __syncthreads();

  int cur = 0;
  for (int kc = 0; kc < 1024; kc += 32) {
    const bool more = (kc + 32) < 1024;
    unsigned short* lAc = sm + cur * A_BUF;
    unsigned short* lBc = sm + B_OFF + cur * B_BUF;
    if (more) {
      const unsigned short* as = aptr + (kc + 32);
      const unsigned short* bs = bptr + (size_t)(kc + 32) * 1024;
      unsigned short* ad = lAs + (cur ^ 1) * A_BUF;
      unsigned short* bd = lBs + (cur ^ 1) * B_BUF;
      async_b128(ad, as);     async_b128(ad + 8, as + 8);
      async_b128(bd, bs);     async_b128(bd + 8, bs + 8);
      if (kc + 64 < 1024) {
        __builtin_prefetch(aptr + (kc + 64), 0, 0);
        __builtin_prefetch(bptr + (size_t)(kc + 64) * 1024, 0, 0);
      }
    }
    Frag a0, a1;
    {
      const unsigned short* ap = lAc + (wm * 32 + lane15) * LDA_S + hb;
      a0.q2[0] = *(const uint4*)(ap);
      a0.q2[1] = *(const uint4*)(ap + 16);
      const unsigned short* ap2 = ap + 16 * LDA_S;
      a1.q2[0] = *(const uint4*)(ap2);
      a1.q2[1] = *(const uint4*)(ap2 + 16);
    }
#pragma unroll
    for (int t = 0; t < 4; ++t) {
      Frag bfr;
      const unsigned short* bp = lBc + wn * 64 + t * 16 + lane15;
#pragma unroll
      for (int i = 0; i < 8; ++i) {
        bfr.u[i]     = bp[(hb + i) * LDB_S];
        bfr.u[8 + i] = bp[(hb + 16 + i) * LDB_S];
      }
      acc[0][t] = WMMA_BF16(a0.v, bfr.v, acc[0][t]);
      acc[1][t] = WMMA_BF16(a1.v, bfr.v, acc[1][t]);
    }
    if (more) wait_async();
    __syncthreads();
    cur ^= 1;
  }

  float bias[4];
#pragma unroll
  for (int t = 0; t < 4; ++t) bias[t] = bo[ntile + wn * 64 + t * 16 + lane15];
#pragma unroll
  for (int s = 0; s < 2; ++s)
#pragma unroll
    for (int r = 0; r < 8; ++r) {
      int m = mtile + wm * 32 + s * 16 + r + hb;
      size_t base = (size_t)m * 1024 + ntile + wn * 64;
#pragma unroll
      for (int t = 0; t < 4; ++t)
        out[base + t * 16 + lane15] = acc[s][t][r] + bias[t];
    }
}

extern "C" void kernel_launch(void* const* d_in, const int* in_sizes, int n_in,
                              void* d_out, int out_size, void* d_ws, size_t ws_size,
                              hipStream_t stream) {
  (void)in_sizes; (void)n_in; (void)out_size; (void)ws_size;
  const float* x  = (const float*)d_in[0];
  const float* Wq = (const float*)d_in[1];
  const float* Wk = (const float*)d_in[2];
  const float* Wv = (const float*)d_in[3];
  const float* Wo = (const float*)d_in[4];
  const float* bo = (const float*)d_in[5];

  const size_t XSEG = (size_t)16384 * 1024;  // u16 elements
  const size_t WSEG = (size_t)1024 * 1024;
  unsigned short* xb  = (unsigned short*)d_ws;
  unsigned short* wqb = xb + XSEG;
  unsigned short* wkb = wqb + WSEG;
  unsigned short* wvb = wkb + WSEG;
  unsigned short* wob = wvb + WSEG;
  unsigned short* qb  = wob + WSEG;
  unsigned short* kb  = qb + XSEG;
  unsigned short* vb  = kb + XSEG;
  unsigned short* ab  = vb + XSEG;
  float* out = (float*)d_out;

  cvt_bf16_kernel<<<dim3(8192), dim3(256), 0, stream>>>(x, xb, 2097152);
  cvt_bf16_kernel<<<dim3(512), dim3(256), 0, stream>>>(Wq, wqb, 131072);
  cvt_bf16_kernel<<<dim3(512), dim3(256), 0, stream>>>(Wk, wkb, 131072);
  cvt_bf16_kernel<<<dim3(512), dim3(256), 0, stream>>>(Wv, wvb, 131072);
  cvt_bf16_kernel<<<dim3(512), dim3(256), 0, stream>>>(Wo, wob, 131072);

  qkv_gemm_kernel<<<dim3(128, 8, 3), dim3(256), 0, stream>>>(xb, wqb, wkb, wvb, qb, kb, vb);
  attn_kernel<<<dim3(64), dim3(128), 0, stream>>>(qb, kb, vb, ab);
  out_gemm_kernel<<<dim3(128, 8), dim3(256), 0, stream>>>(ab, wob, bo, out);
}
